// LIF_hh_neuron_30013231464743
// MI455X (gfx1250) — matrix operand order repaired and checked by reading, not compiled, not run on hardware
//
#include <hip/hip_runtime.h>

#define T_STEPS 15
#define B_DIM   16
#define CIN     64
#define COUT    128
#define H_DIM   56
#define W_DIM   56
#define HW      (H_DIM * W_DIM)   // 3136
#define KDIM    (CIN * 9)         // 576
#define NTILE   32                // spatial positions per block (2 WMMA N-subtiles)
#define DECAY_C  0.2f
#define THRESH_C 0.8f

typedef __attribute__((ext_vector_type(16))) __bf16 v16bf;
typedef __attribute__((ext_vector_type(8)))  float  v8f;

union Frag32 {
    uint4 q[2];   // 32 bytes
    v16bf v;      // 16 bf16 = 32 bytes
};

__device__ __forceinline__ unsigned short f2bf(float f) {
    unsigned int u = __float_as_uint(f);
    u += 0x7FFFu + ((u >> 16) & 1u);       // round-to-nearest-even
    return (unsigned short)(u >> 16);
}

// Pack conv weights fp32 [Cout,Cin,3,3] -> bf16 [Cout][K] (flat layout is identical).
__global__ void pack_weights_kernel(const float* __restrict__ w,
                                    unsigned short* __restrict__ out, int n) {
    int i = blockIdx.x * 256 + threadIdx.x;
    if (i < n) out[i] = f2bf(w[i]);
}

// Fused implicit-GEMM conv (bf16 WMMA) + LIF/HH membrane update, one timestep.
// Grid: (HW/NTILE, B). Block: 256 threads = 8 waves; wave w owns Cout rows [16w,16w+16)
// and computes TWO 16x16 N-subtiles per K-step, reusing one A fragment.
__global__ __launch_bounds__(256)
void conv_lif_step_kernel(const float* __restrict__ x,            // [B,T,CIN,H,W]
                          const unsigned short* __restrict__ wBf, // [COUT][KDIM] bf16
                          const float* __restrict__ convb,        // [COUT]
                          const float* __restrict__ fcw,          // [3]
                          const float* __restrict__ fcb,          // [1]
                          float4* __restrict__ mem,               // [B,COUT,HW] float4
                          float* __restrict__ spike,              // [B,COUT,HW]
                          int t) {
    __shared__ unsigned short sPatch[NTILE * KDIM];  // 36864 B: 32 positions x 576 K

    const int tile = blockIdx.x;   // 0..97, 32 consecutive flat spatial positions
    const int b    = blockIdx.y;
    const int tid  = threadIdx.x;

    // ---- Stage input patches (im2col columns) to LDS as bf16, once per block ----
    const float* xb = x + (size_t)(b * T_STEPS + t) * CIN * HW;
    for (int idx = tid; idx < NTILE * KDIM; idx += 256) {
        int n  = idx / KDIM;
        int k  = idx - n * KDIM;
        int ci = k / 9;
        int r  = k - ci * 9;
        int kh = r / 3;
        int kw = r - kh * 3;
        int p  = tile * NTILE + n;
        int h  = p / W_DIM + kh - 1;
        int w  = p % W_DIM + kw - 1;
        float v = 0.0f;
        if (h >= 0 && h < H_DIM && w >= 0 && w < W_DIM)
            v = xb[(size_t)ci * HW + h * W_DIM + w];
        sPatch[idx] = f2bf(v);
    }
    __syncthreads();

    const int lane = tid & 31;
    const int wv   = tid >> 5;     // wave id: Cout block
    const int hi   = lane >> 4;    // 0/1 half-wave
    const int mlo  = lane & 15;

    const unsigned short* wrow = wBf + (size_t)(wv * 16 + mlo) * KDIM;
    __builtin_prefetch(wrow, 0, 1);  // global_prefetch_b8

    v8f acc0 = {};
    v8f acc1 = {};
    #pragma unroll 3
    for (int kk = 0; kk < KDIM; kk += 32) {
        Frag32 a, b0, b1;
        // A fragment (16x32 bf16): lane holds contiguous k = kk+16hi .. +15
        const uint4* wq = reinterpret_cast<const uint4*>(wrow + kk + hi * 16);
        a.q[0] = wq[0];
        a.q[1] = wq[1];
        // B fragments (32x16 bf16): lane holds column n, contiguous k = kk+16hi .. +15
        const uint4* c0 = reinterpret_cast<const uint4*>(
            sPatch + (size_t)mlo * KDIM + kk + hi * 16);
        const uint4* c1 = reinterpret_cast<const uint4*>(
            sPatch + (size_t)(16 + mlo) * KDIM + kk + hi * 16);
        b0.q[0] = c0[0];
        b0.q[1] = c0[1];
        b1.q[0] = c1[0];
        b1.q[1] = c1[1];
        // Two back-to-back WMMAs sharing the A operand registers
        acc0 = __builtin_amdgcn_wmma_f32_16x16x32_bf16(
            false, a.v, false, b0.v, (short)0, acc0, false, false);
        acc1 = __builtin_amdgcn_wmma_f32_16x16x32_bf16(
            false, a.v, false, b1.v, (short)0, acc1, false, false);
    }

    // ---- Fused LIF/HH epilogue: D layout => lane(mlo)=n, VGPR r => m = 8*hi + r ----
    const float fw0 = fcw[0], fw1 = fcw[1], fw2 = fcw[2], fb = fcb[0];
    #pragma unroll
    for (int st = 0; st < 2; ++st) {
        const int p = tile * NTILE + st * 16 + mlo;
        #pragma unroll
        for (int r = 0; r < 8; ++r) {
            int co = wv * 16 + hi * 8 + r;
            float cv = (st == 0 ? acc0[r] : acc1[r]) + convb[co];
            size_t ei = (size_t)(b * COUT + co) * HW + p;
            float4 mv = mem[ei];
            float sp  = spike[ei];
            float x4  = fw0 * mv.x + fw1 * mv.y + fw2 * mv.z + fb;  // fc on OLD mem[0:3]
            float dec = DECAY_C * (1.0f - sp);
            float n0 = mv.x * dec + cv;
            float n1 = mv.y * dec + cv;
            float n2 = mv.z * dec + cv;
            float n3 = mv.w * dec + x4;
            float m1 = fw0 * n0 + fw1 * n1 + fw2 * n2 + fb + n3;
            mem[ei]   = make_float4(n0, n1, n2, n3);
            spike[ei] = (m1 - THRESH_C) > 0.0f ? 1.0f : 0.0f;       // step activation
        }
    }
}

// 2x2 average pool of spike map -> d_out[:, t]
__global__ void pool_step_kernel(const float* __restrict__ spike,
                                 float* __restrict__ out, int t) {
    const int total = B_DIM * COUT * 28 * 28;
    int idx = blockIdx.x * 256 + threadIdx.x;
    if (idx >= total) return;
    int pw = idx % 28;
    int ph = (idx / 28) % 28;
    int co = (idx / 784) % COUT;
    int b  = idx / (784 * COUT);
    size_t sb = (size_t)(b * COUT + co) * HW;
    int h = ph * 2, w = pw * 2;
    float s = spike[sb + h * W_DIM + w]       + spike[sb + h * W_DIM + w + 1]
            + spike[sb + (h + 1) * W_DIM + w] + spike[sb + (h + 1) * W_DIM + w + 1];
    out[(((size_t)(b * T_STEPS + t) * COUT + co) * 28 + ph) * 28 + pw] = 0.25f * s;
}

extern "C" void kernel_launch(void* const* d_in, const int* in_sizes, int n_in,
                              void* d_out, int out_size, void* d_ws, size_t ws_size,
                              hipStream_t stream) {
    const float* x     = (const float*)d_in[0];   // [16,15,64,56,56]
    const float* convw = (const float*)d_in[1];   // [128,64,3,3]
    const float* convb = (const float*)d_in[2];   // [128]
    const float* fcw   = (const float*)d_in[3];   // [1,3]
    const float* fcb   = (const float*)d_in[4];   // [1]
    float* out = (float*)d_out;                   // [16,15,128,28,28]

    // Workspace layout: bf16 weights | mem state (float4) | spike map
    char* ws = (char*)d_ws;
    unsigned short* wBf = (unsigned short*)ws;                          // 147456 B
    size_t off = (size_t)COUT * KDIM * sizeof(unsigned short);
    float4* mem = (float4*)(ws + off);
    size_t memBytes = (size_t)B_DIM * COUT * HW * 4 * sizeof(float);    // ~103 MB
    off += memBytes;
    float* spike = (float*)(ws + off);
    size_t spkBytes = (size_t)B_DIM * COUT * HW * sizeof(float);        // ~26 MB

    // State must start at zero every call (deterministic, replay-safe)
    hipMemsetAsync(mem, 0, memBytes, stream);
    hipMemsetAsync(spike, 0, spkBytes, stream);

    pack_weights_kernel<<<(COUT * KDIM + 255) / 256, 256, 0, stream>>>(
        convw, wBf, COUT * KDIM);

    dim3 grid(HW / NTILE, B_DIM);   // 98 x 16 blocks
    const int poolBlocks = (B_DIM * COUT * 28 * 28 + 255) / 256;
    for (int t = 0; t < T_STEPS; ++t) {
        conv_lif_step_kernel<<<grid, 256, 0, stream>>>(
            x, wBf, convb, fcw, fcb, mem, spike, t);
        pool_step_kernel<<<poolBlocks, 256, 0, stream>>>(spike, out, t);
    }
}
